// MultiVectorQuantizer_55903294324895
// MI455X (gfx1250) — matrix operand order repaired and checked by reading, not compiled, not run on hardware
//
#include <hip/hip_runtime.h>
#include <hip/hip_bf16.h>

typedef __attribute__((ext_vector_type(16))) _Float16 v16h;
typedef __attribute__((ext_vector_type(8)))  _Float16 v8h;
typedef __attribute__((ext_vector_type(8)))  float    v8f;

#define EMB_DIM     64
#define NUM_EMB     4096
#define N_ROWS      65536              // 8192*512 / 64
#define TOTAL_ELEMS (8192 * 512)
#define NUM_BLOCKS_MAIN 1024           // 64 rows per block (4 waves x 16 rows)

// workspace layout (bytes)
#define WS_EMB_F16_OFF  0
#define WS_EMB_NORM_OFF (NUM_EMB * EMB_DIM * 2)            // 524288
#define WS_PARTIAL_OFF  (WS_EMB_NORM_OFF + NUM_EMB * 4)    // 540672

// ---------------------------------------------------------------------------
// Kernel 1: codebook prep — f32 -> f16 copy + exact f32 row norms ||w||^2
// one wave per codebook row
// ---------------------------------------------------------------------------
__global__ __launch_bounds__(32)
void vq_prep(const float* __restrict__ emb,
             _Float16* __restrict__ emb_h,
             float* __restrict__ emb_norm) {
    int row = blockIdx.x;
    int t   = threadIdx.x;           // 0..31
    const float* er = emb + (size_t)row * EMB_DIM;
    float e0 = er[t];
    float e1 = er[t + 32];
    _Float16* eh = emb_h + (size_t)row * EMB_DIM;
    eh[t]      = (_Float16)e0;
    eh[t + 32] = (_Float16)e1;
    float s = e0 * e0 + e1 * e1;
    #pragma unroll
    for (int m = 16; m >= 1; m >>= 1) s += __shfl_xor(s, m, 32);
    if (t == 0) emb_norm[row] = s;
}

// ---------------------------------------------------------------------------
// Kernel 2: fused distance-GEMM (WMMA f16 -> f32) + argmin + gather + loss
// 128 threads = 4 waves; each wave owns 16 x-rows, scans all 4096 codes.
// kt loop unrolled 2x: independent WMMA chains pipeline across tiles and
// hide the WMMA->VALU RAW hazard (4 coexec slots) + load waits.
// ---------------------------------------------------------------------------
__global__ __launch_bounds__(128)
void vq_main(const float* __restrict__ x,
             const float* __restrict__ emb,
             const _Float16* __restrict__ emb_h,
             const float* __restrict__ emb_norm,
             float* __restrict__ out,
             float* __restrict__ partial) {
    __shared__ float s_loss[4];

    const int wave    = threadIdx.x >> 5;
    const int lane    = threadIdx.x & 31;
    const int rowBase = blockIdx.x * 64 + wave * 16;
    const int m       = lane & 15;     // A-row / B-col / C-col owned by lane
    const int kg      = lane >> 4;     // lane half selects K sub-block

    // ---- load A fragments: 16 rows of x, K=0..63, contiguous K per lane half ----
    // frag element i (i<8):  K = kg*16 + i        (VGPR0..3)
    // frag element i (i>=8): K = kg*16 + 8 + i-8 (VGPR4..7)
    const float* xr = x + (size_t)(rowBase + m) * EMB_DIM;
    v16h a0, a1;
    {
        const int k0 = kg * 16;
        const int k1 = kg * 16 + 8;
        #pragma unroll
        for (int i = 0; i < 8; ++i) {
            a0[i]     = (_Float16)xr[k0 + i];
            a0[8 + i] = (_Float16)xr[k1 + i];
            a1[i]     = (_Float16)xr[32 + k0 + i];
            a1[8 + i] = (_Float16)xr[32 + k1 + i];
        }
    }

    float minv[8];
    int   mini[8];
    #pragma unroll
    for (int r = 0; r < 8; ++r) { minv[r] = 3.4e38f; mini[r] = 0; }

    const int n = m;                   // C/D column owned by this lane
    // Each codebook row is 64 f16 = 4 chunks of 32B (v16h).
    const v16h* ebase = (const v16h*)emb_h;

    #pragma unroll 2
    for (int kt = 0; kt < NUM_EMB / 16; ++kt) {
        // ---- B fragments: contiguous K order as A: lanes 0-15 hold K=0..15,
        // lanes 16-31 hold K=16..31 -> 16 contiguous f16 per lane per frag.
        const v16h* er16 = ebase + (size_t)(kt * 16 + n) * 4;
        v16h b0 = er16[kg];                          // K = kg*16 ..  +15
        v16h b1 = er16[2 + kg];                      // K = 32+kg*16 .. +15
        float nn = emb_norm[kt * 16 + n];

        if (kt + 2 < NUM_EMB / 16)                   // global_prefetch_b8
            __builtin_prefetch(emb_h + (size_t)((kt + 2) * 16 + n) * EMB_DIM, 0, 3);

        v8f c = {};
        c = __builtin_amdgcn_wmma_f32_16x16x32_f16(false, a0, false, b0,
                                                   (short)0, c, false, false);
        c = __builtin_amdgcn_wmma_f32_16x16x32_f16(false, a1, false, b1,
                                                   (short)0, c, false, false);

        const int idx = kt * 16 + n;
        #pragma unroll
        for (int r = 0; r < 8; ++r) {
            float d = fmaf(-2.0f, c[r], nn);         // ||w||^2 - 2 x.w
            if (d < minv[r]) { minv[r] = d; mini[r] = idx; }
        }
    }

    // ---- cross-lane argmin within each 16-lane half (rows r / 8+r) ----
    #pragma unroll
    for (int msk = 1; msk < 16; msk <<= 1) {
        #pragma unroll
        for (int r = 0; r < 8; ++r) {
            float ov = __shfl_xor(minv[r], msk, 32);
            int   oi = __shfl_xor(mini[r], msk, 32);
            if (ov < minv[r] || (ov == minv[r] && oi < mini[r])) {
                minv[r] = ov; mini[r] = oi;
            }
        }
    }

    // ---- gather winning codebook rows (f32), emit quantized, loss ----
    float lsum = 0.0f;
    #pragma unroll
    for (int r = 0; r < 8; ++r) {
        const int row = rowBase + r + 8 * kg;        // C VGPR r -> row r / 8+r
        const int idx = mini[r];                     // uniform across the half
        const float4* e4 = (const float4*)(emb + (size_t)idx * EMB_DIM);
        const float4* x4 = (const float4*)(x   + (size_t)row * EMB_DIM);
        float4*       o4 = (float4*)(out + (size_t)row * EMB_DIM);
        float4 e  = e4[n];
        float4 xv = x4[n];
        o4[n] = e;                                   // quantized_out == quantized
        float dx = e.x - xv.x, dy = e.y - xv.y;
        float dz = e.z - xv.z, dw = e.w - xv.w;
        lsum += dx * dx + dy * dy + dz * dz + dw * dw;
    }
    #pragma unroll
    for (int msk = 16; msk >= 1; msk >>= 1) lsum += __shfl_xor(lsum, msk, 32);
    if (lane == 0) s_loss[wave] = lsum;
    __syncthreads();
    if (threadIdx.x == 0)
        partial[blockIdx.x] = s_loss[0] + s_loss[1] + s_loss[2] + s_loss[3];
}

// ---------------------------------------------------------------------------
// Kernel 3: deterministic final loss reduction
// loss = (1.0 + 0.25) * mean((q - x)^2)
// ---------------------------------------------------------------------------
__global__ __launch_bounds__(256)
void vq_loss(const float* __restrict__ partial, float* __restrict__ loss_out) {
    __shared__ float s[256];
    int t = threadIdx.x;
    float v = 0.0f;
    for (int i = t; i < NUM_BLOCKS_MAIN; i += 256) v += partial[i];
    s[t] = v;
    __syncthreads();
    for (int stride = 128; stride > 0; stride >>= 1) {
        if (t < stride) s[t] += s[t + stride];
        __syncthreads();
    }
    if (t == 0) loss_out[0] = s[0] * (1.25f / (float)TOTAL_ELEMS);
}

extern "C" void kernel_launch(void* const* d_in, const int* in_sizes, int n_in,
                              void* d_out, int out_size, void* d_ws, size_t ws_size,
                              hipStream_t stream) {
    (void)in_sizes; (void)n_in; (void)out_size; (void)ws_size;
    const float* x   = (const float*)d_in[0];   // [8192, 512] f32
    const float* emb = (const float*)d_in[1];   // [4096, 64]  f32
    float* out = (float*)d_out;                 // [8192*512] quantized + [1] loss

    char* ws = (char*)d_ws;
    _Float16* emb_h    = (_Float16*)(ws + WS_EMB_F16_OFF);
    float*    emb_norm = (float*)(ws + WS_EMB_NORM_OFF);
    float*    partial  = (float*)(ws + WS_PARTIAL_OFF);

    vq_prep<<<NUM_EMB, 32, 0, stream>>>(emb, emb_h, emb_norm);
    vq_main<<<NUM_BLOCKS_MAIN, 128, 0, stream>>>(x, emb, emb_h, emb_norm,
                                                 out, partial);
    vq_loss<<<1, 256, 0, stream>>>(partial, out + (size_t)TOTAL_ELEMS);
}
